// GAT_20255065768645
// MI455X (gfx1250) — compile-verified, matrix operand order checked by reading, not run on hardware
//
#include <hip/hip_runtime.h>

// ---------------------------------------------------------------------------
// Types for CDNA5 WMMA (wave32): v_wmma_f32_16x16x32_bf16
// ---------------------------------------------------------------------------
typedef __attribute__((ext_vector_type(16))) __bf16        v16bf;
typedef __attribute__((ext_vector_type(8)))  float         v8f;
typedef __attribute__((ext_vector_type(4)))  unsigned int  v4u;
typedef __attribute__((ext_vector_type(8)))  int           v8i;
typedef __attribute__((ext_vector_type(4)))  int           v4i;

union FragAB { v16bf v; v4u q[2]; };   // 32 bytes = 8 VGPRs (16-bit A/B operand)

__device__ __forceinline__ unsigned short f32_to_bf16(float f) {
  unsigned int u = __float_as_uint(f);
  u += 0x7FFFu + ((u >> 16) & 1u);     // round-to-nearest-even
  return (unsigned short)(u >> 16);
}

// ---------------------------------------------------------------------------
// Tensor Data Mover: stage a [rows x cols] bf16 panel (row stride = stride
// elements) from global into LDS at byte offset lds_byte. 2D descriptor,
// Group0/Group1 packed per CDNA5 ISA 8.3/8.4; groups 2/3 zero (2D tensor).
// Toolchain here exposes the 6-arg builtin: (g0, g1, g2, g3, g2b, cpol).
// ---------------------------------------------------------------------------
#if defined(__gfx1250__) && __has_builtin(__builtin_amdgcn_tensor_load_to_lds) && \
    __has_builtin(__builtin_amdgcn_s_wait_tensorcnt)
#define USE_TDM 1
__device__ __forceinline__ void tdm_load_panel(unsigned lds_byte,
                                               const unsigned short* gptr,
                                               unsigned rows, unsigned cols,
                                               unsigned stride) {
  unsigned long long ga = (unsigned long long)gptr;
  v4u g0;
  g0[0] = 1u;                                    // count=1, user descriptor
  g0[1] = lds_byte;                              // lds_addr        [63:32]
  g0[2] = (unsigned)(ga & 0xffffffffu);          // global_addr lo  [95:64]
  g0[3] = (unsigned)((ga >> 32) & 0x01ffffffu)   // global_addr hi  [120:96]
        | (2u << 30);                            // type=2 (image)  [127:126]
  v8i g1;
  g1[0] = (int)(1u << 16);                       // data_size=1 (2B); wg_mask=0
  g1[1] = (int)((cols & 0xffffu) << 16);         // tensor_dim0 lo  [63:48]
  g1[2] = (int)(((cols >> 16) & 0xffffu)         // tensor_dim0 hi  [79:64]
        | ((rows & 0xffffu) << 16));             // tensor_dim1 lo  [95:80]
  g1[3] = (int)((cols & 0xffffu) << 16);         // tile_dim0       [127:112]
  g1[4] = (int)(rows & 0xffffu);                 // tile_dim1       [143:128]
  g1[5] = (int)stride;                           // tensor_dim0_stride lo
  g1[6] = 0;                                     // stride hi / dim1_stride
  g1[7] = 0;
  v4i z4 = {0, 0, 0, 0};
  v8i z8 = {0, 0, 0, 0, 0, 0, 0, 0};
  __builtin_amdgcn_tensor_load_to_lds(g0, g1, z4, z4, z8, 0);
}
#else
#define USE_TDM 0
#endif

// ---------------------------------------------------------------------------
// Conversion / fill kernels
// ---------------------------------------------------------------------------
__global__ void k_pad_bf16(const float* __restrict__ src, unsigned short* __restrict__ dst,
                           int rows, int kin, int kpad) {
  long t = (long)blockIdx.x * blockDim.x + threadIdx.x;
  if (t >= (long)rows * kpad) return;
  int  k = (int)(t % kpad);
  long r = t / kpad;
  float v = (k < kin) ? src[r * (long)kin + k] : 0.f;
  dst[t] = f32_to_bf16(v);
}

// W [kin, nout] fp32 (row-major) -> Wt [nout, kpad] bf16 (rows contiguous in K)
__global__ void k_transpose_bf16(const float* __restrict__ W, unsigned short* __restrict__ Wt,
                                 int kin, int nout, int kpad) {
  long t = (long)blockIdx.x * blockDim.x + threadIdx.x;
  if (t >= (long)nout * kpad) return;
  int k = (int)(t % kpad);
  int n = (int)(t / kpad);
  float v = (k < kin) ? W[(size_t)k * nout + n] : 0.f;
  Wt[t] = f32_to_bf16(v);
}

__global__ void k_fill_f32(float* __restrict__ p, float v, long n) {
  long t = (long)blockIdx.x * blockDim.x + threadIdx.x;
  if (t < n) p[t] = v;
}

// ---------------------------------------------------------------------------
// WMMA GEMM:  C[M,Nout] = A[M,K](bf16) @ Bt[Nout,K](bf16)^T , fp32 accumulate.
// lda == K (contiguous panels). Block = 128 threads = 4 waves.
// Block tile: 16(M) x 256(N); each wave owns 16x64 = four 16x16 accumulators.
// Full 16xK A panel staged once into LDS (TDM if available), so the K-loop
// has no barriers: 1 A-fragment ds_load feeds 4 WMMAs.
// 16-bit A/B operand layout per lane L: row = L%16,
//   elems 0..7 = K[(L/16)*8 .. +8), elems 8..15 = K[(L/16)*8+16 .. +8).
// ---------------------------------------------------------------------------
#define TILE_K 32
#define MAX_K  800

__global__ __launch_bounds__(128)
void k_gemm_bf16_wmma(const unsigned short* __restrict__ A,
                      const unsigned short* __restrict__ Bt,
                      float* __restrict__ C,
                      int M, int Nout, int K) {
  __shared__ __align__(16) unsigned short sA[16 * MAX_K];   // <= 25.6 KB

  const int tid   = threadIdx.x;
  const int wave  = tid >> 5;
  const int lane  = tid & 31;
  const int lrow  = lane & 15;
  const int lhalf = lane >> 4;
  const int m0    = blockIdx.x * 16;
  const int nw    = blockIdx.y * 256 + wave * 64;

  // ---- stage A panel [16 x K] (contiguous) into LDS, once ----
#if USE_TDM
  if (tid < 32) {                           // wave 0 issues the DMA descriptor
    tdm_load_panel((unsigned)(unsigned long long)&sA[0],
                   A + (size_t)m0 * K, 16u, (unsigned)K, (unsigned)K);
    __builtin_amdgcn_s_wait_tensorcnt(0);
  }
#else
  {
    const v4u* Aq = (const v4u*)(A + (size_t)m0 * K);
    v4u* s4 = (v4u*)sA;
    const int tot = 2 * K;                  // 16*K halves / 8 per v4u
    for (int i = tid; i < tot; i += 128) s4[i] = Aq[i];
  }
#endif
  __syncthreads();

  const int n0 = nw,      n0v = (n0 < Nout) ? n0 : (Nout - 16);
  const int n1 = nw + 16, n1v = (n1 < Nout) ? n1 : (Nout - 16);
  const int n2 = nw + 32, n2v = (n2 < Nout) ? n2 : (Nout - 16);
  const int n3 = nw + 48, n3v = (n3 < Nout) ? n3 : (Nout - 16);

  const unsigned short* sArow = sA + lrow * K + lhalf * 8;
  const unsigned short* b0p = Bt + (size_t)(n0v + lrow) * K + lhalf * 8;
  const unsigned short* b1p = Bt + (size_t)(n1v + lrow) * K + lhalf * 8;
  const unsigned short* b2p = Bt + (size_t)(n2v + lrow) * K + lhalf * 8;
  const unsigned short* b3p = Bt + (size_t)(n3v + lrow) * K + lhalf * 8;

  v8f acc0 = {}, acc1 = {}, acc2 = {}, acc3 = {};

  for (int k0 = 0; k0 < K; k0 += TILE_K) {
    FragAB a;
    const v4u* pa = (const v4u*)(sArow + k0);
    a.q[0] = pa[0];
    a.q[1] = pa[2];                                   // +16 halves

    if (k0 + TILE_K < K)                              // global_prefetch_b8
      __builtin_prefetch(b0p + k0 + TILE_K, 0, 1);

    FragAB b;
    const v4u* pb;
    pb = (const v4u*)(b0p + k0);  b.q[0] = pb[0]; b.q[1] = pb[2];
    acc0 = __builtin_amdgcn_wmma_f32_16x16x32_bf16(false, a.v, false, b.v,
                                                   (short)0, acc0, false, false);
    pb = (const v4u*)(b1p + k0);  b.q[0] = pb[0]; b.q[1] = pb[2];
    acc1 = __builtin_amdgcn_wmma_f32_16x16x32_bf16(false, a.v, false, b.v,
                                                   (short)0, acc1, false, false);
    pb = (const v4u*)(b2p + k0);  b.q[0] = pb[0]; b.q[1] = pb[2];
    acc2 = __builtin_amdgcn_wmma_f32_16x16x32_bf16(false, a.v, false, b.v,
                                                   (short)0, acc2, false, false);
    pb = (const v4u*)(b3p + k0);  b.q[0] = pb[0]; b.q[1] = pb[2];
    acc3 = __builtin_amdgcn_wmma_f32_16x16x32_bf16(false, a.v, false, b.v,
                                                   (short)0, acc3, false, false);
  }

  // C/D layout: VGPR r -> M = r + 8*(lane>=16), N = lane%16
  const int row = m0 + (lhalf << 3);
#pragma unroll
  for (int r = 0; r < 8; ++r) {
    if (n0 < Nout) C[(size_t)(row + r) * Nout + n0 + lrow] = acc0[r];
    if (n1 < Nout) C[(size_t)(row + r) * Nout + n1 + lrow] = acc1[r];
    if (n2 < Nout) C[(size_t)(row + r) * Nout + n2 + lrow] = acc2[r];
    if (n3 < Nout) C[(size_t)(row + r) * Nout + n3 + lrow] = acc3[r];
  }
}

// ---------------------------------------------------------------------------
// Attention kernels
// ---------------------------------------------------------------------------
__global__ void k_alphas(const float* __restrict__ h, const float* __restrict__ a_s,
                         const float* __restrict__ a_d, float* __restrict__ als,
                         float* __restrict__ ald, int N, int H, int Cc) {
  int t = blockIdx.x * blockDim.x + threadIdx.x;
  if (t >= N * H) return;
  int n = t / H, hh = t - n * H;
  const float* hp = h + (size_t)n * H * Cc + (size_t)hh * Cc;
  const float* ps = a_s + (size_t)hh * Cc;
  const float* pd = a_d + (size_t)hh * Cc;
  float s = 0.f, d = 0.f;
  for (int c = 0; c < Cc; ++c) { float v = hp[c]; s += v * ps[c]; d += v * pd[c]; }
  als[t] = s; ald[t] = d;
}

__global__ void k_edge_scores_max(const int* __restrict__ ei,
                                  const float* __restrict__ als, const float* __restrict__ ald,
                                  float* __restrict__ esc, float* __restrict__ mx,
                                  int E, int E2, int H) {
  long t = (long)blockIdx.x * blockDim.x + threadIdx.x;
  if (t >= (long)E2 * H) return;
  int e  = (int)(t / H);
  int hh = (int)(t - (long)e * H);
  int s  = (e < E) ? ei[e] : (e - E);
  int d  = (e < E) ? ei[(size_t)E + e] : (e - E);
  float v = als[(size_t)s * H + hh] + ald[(size_t)d * H + hh];
  v = (v > 0.f) ? v : 0.2f * v;
  esc[t] = v;
  atomicMax(&mx[(size_t)d * H + hh], v);       // global_atomic_max_num_f32
}

__global__ void k_edge_expsum(const int* __restrict__ ei,
                              float* __restrict__ esc, const float* __restrict__ mx,
                              float* __restrict__ den, int E, int E2, int H) {
  long t = (long)blockIdx.x * blockDim.x + threadIdx.x;
  if (t >= (long)E2 * H) return;
  int e  = (int)(t / H);
  int hh = (int)(t - (long)e * H);
  int d  = (e < E) ? ei[(size_t)E + e] : (e - E);
  float ex = __expf(esc[t] - mx[(size_t)d * H + hh]);
  esc[t] = ex;
  atomicAdd(&den[(size_t)d * H + hh], ex);
}

__global__ void k_edge_aggregate(const int* __restrict__ ei,
                                 const float* __restrict__ esc, const float* __restrict__ den,
                                 const float* __restrict__ hsrc, float* __restrict__ agg,
                                 int E, int E2, int H, int Cc, int F) {
  long t = (long)blockIdx.x * blockDim.x + threadIdx.x;
  if (t >= (long)E2 * F) return;
  int e   = (int)(t / F);
  int idx = (int)(t - (long)e * F);
  int hh  = idx / Cc;
  int s   = (e < E) ? ei[e] : (e - E);
  int d   = (e < E) ? ei[(size_t)E + e] : (e - E);
  float alpha = esc[(size_t)e * H + hh] / (den[(size_t)d * H + hh] + 1e-16f);
  atomicAdd(&agg[(size_t)d * F + idx], hsrc[(size_t)s * F + idx] * alpha);
}

__global__ void k_bias_relu_bf16(const float* __restrict__ agg, const float* __restrict__ b,
                                 unsigned short* __restrict__ xout, int N, int F) {
  long t = (long)blockIdx.x * blockDim.x + threadIdx.x;
  if (t >= (long)N * F) return;
  int f = (int)(t % F);
  float v = agg[t] + b[f];
  xout[t] = f32_to_bf16(v > 0.f ? v : 0.f);
}

// ---------------------------------------------------------------------------
// Mean pooling
// ---------------------------------------------------------------------------
__global__ void k_pool_accum(const float* __restrict__ agg, const float* __restrict__ b,
                             const int* __restrict__ batch, float* __restrict__ sums,
                             float* __restrict__ cnt, int N, int F) {
  long t = (long)blockIdx.x * blockDim.x + threadIdx.x;
  if (t >= (long)N * F) return;
  int n = (int)(t / F);
  int f = (int)(t - (long)n * F);
  float v = agg[t] + b[f];
  int g = batch[n];
  atomicAdd(&sums[(size_t)g * F + f], v);
  if (f == 0) atomicAdd(&cnt[g], 1.0f);
}

__global__ void k_pool_finalize(const float* __restrict__ sums, const float* __restrict__ cnt,
                                float* __restrict__ out, int G, int F) {
  int t = blockIdx.x * blockDim.x + threadIdx.x;
  if (t >= G * F) return;
  int g = t / F;
  out[t] = sums[t] / fmaxf(cnt[g], 1.0f);
}

// ---------------------------------------------------------------------------
// Driver
// ---------------------------------------------------------------------------
static void run_gat_layer(const unsigned short* Abf, const unsigned short* Wt,
                          int Kpad, int F, int H, int Cc,
                          const float* a_s, const float* a_d, const int* ei,
                          float* hbuf, float* als, float* ald, float* mx, float* den,
                          float* esc, float* agg, hipStream_t stream) {
  constexpr int N = 50000, E = 800000, E2 = 850000;
  dim3 gg((unsigned)(N / 16), (unsigned)((F + 255) / 256));
  k_gemm_bf16_wmma<<<gg, 128, 0, stream>>>(Abf, Wt, hbuf, N, F, Kpad);

  int NH = N * H;
  k_alphas<<<(NH + 255) / 256, 256, 0, stream>>>(hbuf, a_s, a_d, als, ald, N, H, Cc);
  k_fill_f32<<<(NH + 255) / 256, 256, 0, stream>>>(mx, -3.0e38f, NH);
  k_fill_f32<<<(NH + 255) / 256, 256, 0, stream>>>(den, 0.f, NH);
  long NF = (long)N * F;
  k_fill_f32<<<(unsigned)((NF + 255) / 256), 256, 0, stream>>>(agg, 0.f, NF);

  long EH = (long)E2 * H;
  unsigned gEH = (unsigned)((EH + 255) / 256);
  k_edge_scores_max<<<gEH, 256, 0, stream>>>(ei, als, ald, esc, mx, E, E2, H);
  k_edge_expsum<<<gEH, 256, 0, stream>>>(ei, esc, mx, den, E, E2, H);
  long EF = (long)E2 * F;
  k_edge_aggregate<<<(unsigned)((EF + 255) / 256), 256, 0, stream>>>(
      ei, esc, den, hbuf, agg, E, E2, H, Cc, F);
}

extern "C" void kernel_launch(void* const* d_in, const int* in_sizes, int n_in,
                              void* d_out, int out_size, void* d_ws, size_t ws_size,
                              hipStream_t stream) {
  const float* x     = (const float*)d_in[0];
  const int*   ei    = (const int*)d_in[1];
  const int*   batch = (const int*)d_in[2];
  const float* W1    = (const float*)d_in[3];
  const float* as1   = (const float*)d_in[4];
  const float* ad1   = (const float*)d_in[5];
  const float* b1    = (const float*)d_in[6];
  const float* W2    = (const float*)d_in[7];
  const float* as2   = (const float*)d_in[8];
  const float* ad2   = (const float*)d_in[9];
  const float* b2    = (const float*)d_in[10];
  const float* W3    = (const float*)d_in[11];
  const float* as3   = (const float*)d_in[12];
  const float* ad3   = (const float*)d_in[13];
  const float* b3    = (const float*)d_in[14];

  constexpr int N = 50000, E2 = 850000, G = 512;
  constexpr int IN = 775, KP1 = 800, HC = 256, OUTF = 32;

  // ---- carve workspace ----
  char* ws = (char*)d_ws;
  size_t off = 0;
  auto alloc = [&](size_t bytes) -> void* {
    void* p = ws + off;
    off += (bytes + 255) & ~(size_t)255;
    return p;
  };
  unsigned short* xbf  = (unsigned short*)alloc((size_t)N * KP1 * 2);  // reused each layer
  unsigned short* wt1  = (unsigned short*)alloc((size_t)HC * KP1 * 2);
  unsigned short* wt2  = (unsigned short*)alloc((size_t)HC * HC * 2);
  unsigned short* wt3  = (unsigned short*)alloc((size_t)OUTF * HC * 2);
  float* hbuf = (float*)alloc((size_t)N * HC * 4);
  float* agg  = (float*)alloc((size_t)N * HC * 4);
  float* als  = (float*)alloc((size_t)N * 4 * 4);
  float* ald  = (float*)alloc((size_t)N * 4 * 4);
  float* mx   = (float*)alloc((size_t)N * 4 * 4);
  float* den  = (float*)alloc((size_t)N * 4 * 4);
  float* esc  = (float*)alloc((size_t)E2 * 4 * 4);
  float* sums = (float*)alloc((size_t)G * OUTF * 4);
  float* cnt  = (float*)alloc((size_t)G * 4);
  (void)ws_size; (void)in_sizes; (void)n_in; (void)out_size;

  // ---- precision conversion (fp32 -> bf16, K-padded / transposed weights) ----
  long nx = (long)N * KP1;
  k_pad_bf16<<<(unsigned)((nx + 255) / 256), 256, 0, stream>>>(x, xbf, N, IN, KP1);
  long nw1 = (long)HC * KP1;
  k_transpose_bf16<<<(unsigned)((nw1 + 255) / 256), 256, 0, stream>>>(W1, wt1, IN, HC, KP1);
  long nw2 = (long)HC * HC;
  k_transpose_bf16<<<(unsigned)((nw2 + 255) / 256), 256, 0, stream>>>(W2, wt2, HC, HC, HC);
  long nw3 = (long)OUTF * HC;
  k_transpose_bf16<<<(unsigned)((nw3 + 255) / 256), 256, 0, stream>>>(W3, wt3, HC, OUTF, HC);

  // ---- layer 1 (H=4, C=64) ----
  run_gat_layer(xbf, wt1, KP1, HC, 4, 64, as1, ad1, ei,
                hbuf, als, ald, mx, den, esc, agg, stream);
  long nf = (long)N * HC;
  k_bias_relu_bf16<<<(unsigned)((nf + 255) / 256), 256, 0, stream>>>(agg, b1, xbf, N, HC);

  // ---- layer 2 (H=4, C=64) ----
  run_gat_layer(xbf, wt2, HC, HC, 4, 64, as2, ad2, ei,
                hbuf, als, ald, mx, den, esc, agg, stream);
  k_bias_relu_bf16<<<(unsigned)((nf + 255) / 256), 256, 0, stream>>>(agg, b2, xbf, N, HC);

  // ---- layer 3 (H=1, C=32) ----
  run_gat_layer(xbf, wt3, HC, OUTF, 1, 32, as3, ad3, ei,
                hbuf, als, ald, mx, den, esc, agg, stream);

  // ---- global mean pool ----
  k_fill_f32<<<(G * OUTF + 255) / 256, 256, 0, stream>>>(sums, 0.f, (long)G * OUTF);
  k_fill_f32<<<(G + 255) / 256, 256, 0, stream>>>(cnt, 0.f, (long)G);
  long np = (long)N * OUTF;
  k_pool_accum<<<(unsigned)((np + 255) / 256), 256, 0, stream>>>(agg, b3, batch, sums, cnt, N, OUTF);
  k_pool_finalize<<<(G * OUTF + 255) / 256, 256, 0, stream>>>(sums, cnt, (float*)d_out, G, OUTF);
}
